// GroupedQueryAttention_1563368095957
// MI455X (gfx1250) — matrix operand order repaired and checked by reading, not compiled, not run on hardware
//
#include <hip/hip_runtime.h>
#include <stdint.h>

#define B_   2
#define S_   2048
#define D_   2048
#define H_   16
#define KVH_ 4
#define HD_  128
#define GROUPS_ (H_/KVH_)

typedef __bf16 bf16x16 __attribute__((ext_vector_type(16)));
typedef float f32x8   __attribute__((ext_vector_type(8)));
typedef int   v4i_    __attribute__((ext_vector_type(4)));

union Frag { bf16x16 v; unsigned int u[8]; };

// ---- async global->LDS path (CDNA5 GLOBAL_LOAD_ASYNC_TO_LDS_*, ASYNCcnt) ----
#if defined(__gfx1250__) && defined(__AMDGCN__)
#  if __has_builtin(__builtin_amdgcn_global_load_async_to_lds_b128) && \
      __has_builtin(__builtin_amdgcn_s_wait_asynccnt)
#    define USE_ASYNC_LDS 1
#  endif
#endif
#ifndef USE_ASYNC_LDS
#  define USE_ASYNC_LDS 0
#endif

#if USE_ASYNC_LDS
__device__ __forceinline__ void async_copy_b128(const void* g, void* l) {
  // prototype: (global v4i*, local v4i*, imm offset, imm cpol)
  __builtin_amdgcn_global_load_async_to_lds_b128(
      (__attribute__((address_space(1))) v4i_*)g,
      (__attribute__((address_space(3))) v4i_*)l, 0, 0);
}
#endif

__device__ __forceinline__ unsigned short f2bf(float f) {
  unsigned int u = __float_as_uint(f);
  u += 0x7FFFu + ((u >> 16) & 1u);            // round-to-nearest-even
  return (unsigned short)(u >> 16);
}
__device__ __forceinline__ float bf2f(unsigned short h) {
  return __uint_as_float(((unsigned int)h) << 16);
}
// A-matrix (16x32 bf16) K offset per ISA 7.12.2
__device__ __forceinline__ int fragKA(int p, int half) { return ((p >> 2) << 4) + (half << 3) + ((p & 3) << 1); }
// B-matrix (32x16 bf16) K offset (same table order as A)
__device__ __forceinline__ int fragKB(int p, int half) { return ((p >> 2) << 4) + (half << 3) + ((p & 3) << 1); }

// ---------------------------------------------------------------- convert
__global__ void k_f32_to_bf16(const float* __restrict__ src,
                              unsigned short* __restrict__ dst, int n) {
  int i = blockIdx.x * blockDim.x + threadIdx.x;
  if (i < n) dst[i] = f2bf(src[i]);
}

// ---------------------------------------------------------------- GEMM (bf16 -> f32 acc)
// C[M,N] = A[M,K] * Bw[K,N].  mode 0: write bf16 into [B, heads, S, HD] layout.
// mode 1: write f32 row-major (final output projection).
#define GSTR 48
__global__ __launch_bounds__(256) void k_gemm_bf16(
    const unsigned short* __restrict__ A, const unsigned short* __restrict__ Bw,
    void* __restrict__ out, int M, int N, int K, int mode, int heads)
{
  __shared__ unsigned short Alds[128 * GSTR];
  __shared__ unsigned short Blds[128 * GSTR];

  const int tid  = threadIdx.x;
  const int lane = tid & 31;
  const int wave = tid >> 5;
  const int half = lane >> 4;
  const int lcol = lane & 15;
  const int mBlock = blockIdx.y * 128;
  const int nBlock = blockIdx.x * 128;
  const int mW = (wave & 1) * 64;   // 2 waves along M (4 m-tiles of 16 each)
  const int nW = (wave >> 1) * 32;  // 4 waves along N (2 n-tiles of 16 each)

  f32x8 acc[4][2];
  for (int i = 0; i < 4; ++i)
    for (int j = 0; j < 2; ++j)
      for (int e = 0; e < 8; ++e) acc[i][j][e] = 0.0f;

  for (int kb = 0; kb < K; kb += 32) {
    // cooperative A slab: 128 rows x 32 cols (row-major in LDS)
    {
      int r = tid >> 1, c0 = (tid & 1) * 16;
      const unsigned short* gp = &A[(size_t)(mBlock + r) * K + kb + c0];
      unsigned short* lp = &Alds[r * GSTR + c0];
#if USE_ASYNC_LDS
      async_copy_b128(gp, lp);            // 16B, no VGPR roundtrip
      async_copy_b128(gp + 8, lp + 8);
#else
      const uint4* g = (const uint4*)gp;
      uint4 v0 = g[0], v1 = g[1];
      unsigned int* d = (unsigned int*)lp;
      d[0] = v0.x; d[1] = v0.y; d[2] = v0.z; d[3] = v0.w;
      d[4] = v1.x; d[5] = v1.y; d[6] = v1.z; d[7] = v1.w;
#endif
    }
    // cooperative B slab: 32 rows x 128 cols, stored transposed [n][k]
    {
      int kr = tid >> 3, n0 = (tid & 7) * 16;
      const uint4* g = (const uint4*)&Bw[(size_t)(kb + kr) * N + nBlock + n0];
      uint4 v0 = g[0], v1 = g[1];
      const unsigned short* sv = (const unsigned short*)&v0;
      for (int e = 0; e < 8; ++e) Blds[(n0 + e) * GSTR + kr] = sv[e];
      sv = (const unsigned short*)&v1;
      for (int e = 0; e < 8; ++e) Blds[(n0 + 8 + e) * GSTR + kr] = sv[e];
    }
#if USE_ASYNC_LDS
    __builtin_amdgcn_s_wait_asynccnt(0);
#endif
    __syncthreads();
    if (kb + 32 < K) {
      __builtin_prefetch(&A[(size_t)(mBlock + (tid >> 1)) * K + kb + 32], 0, 1);
      __builtin_prefetch(&Bw[(size_t)(kb + 32 + (tid >> 3)) * N + nBlock + (tid & 7) * 16], 0, 1);
    }

    Frag af[4], bfr[2];
    for (int mt = 0; mt < 4; ++mt) {
      const unsigned short* base = &Alds[(mW + mt * 16 + lcol) * GSTR];
      for (int p = 0; p < 8; ++p) af[mt].u[p] = *(const unsigned int*)&base[fragKA(p, half)];
    }
    for (int nt = 0; nt < 2; ++nt) {
      const unsigned short* base = &Blds[(nW + nt * 16 + lcol) * GSTR];
      for (int p = 0; p < 8; ++p) bfr[nt].u[p] = *(const unsigned int*)&base[fragKB(p, half)];
    }
    for (int mt = 0; mt < 4; ++mt)
      for (int nt = 0; nt < 2; ++nt)
        acc[mt][nt] = __builtin_amdgcn_wmma_f32_16x16x32_bf16(
            false, af[mt].v, false, bfr[nt].v, (short)0, acc[mt][nt], false, false);
    __syncthreads();
  }

  // epilogue
  for (int mt = 0; mt < 4; ++mt) {
    for (int nt = 0; nt < 2; ++nt) {
      int gn = nBlock + nW + nt * 16 + lcol;
      for (int i = 0; i < 8; ++i) {
        int gm = mBlock + mW + mt * 16 + i + half * 8;
        float vv = acc[mt][nt][i];
        if (mode == 0) {
          int b = gm >> 11, s = gm & (S_ - 1);
          int h = gn >> 7,  d = gn & (HD_ - 1);
          ((unsigned short*)out)[(((size_t)(b * heads + h)) * S_ + s) * HD_ + d] = f2bf(vv);
        } else {
          ((float*)out)[(size_t)gm * N + gn] = vv;
        }
      }
    }
  }
}

// ---------------------------------------------------------------- RoPE (in place, bf16)
// layout [B, heads, S, HD]; pairs (j, j+64), cos/sin are [S, 64] f32.
__global__ void k_rope(unsigned short* __restrict__ qk, const float* __restrict__ cosp,
                       const float* __restrict__ sinp, int totalRows)
{
  int idx = blockIdx.x * blockDim.x + threadIdx.x;
  if (idx >= totalRows * 64) return;
  int j = idx & 63;
  int row = idx >> 6;
  int s = row & (S_ - 1);
  size_t base = (size_t)row * HD_;
  float v0 = bf2f(qk[base + j]);
  float v1 = bf2f(qk[base + j + 64]);
  float c0 = cosp[s * 64 + (j >> 1)];
  float s0 = sinp[s * 64 + (j >> 1)];
  float c1 = cosp[s * 64 + 32 + (j >> 1)];
  float s1 = sinp[s * 64 + 32 + (j >> 1)];
  qk[base + j]      = f2bf(v0 * c0 - v1 * s0);   // x*cos + (-x_hi)*sin
  qk[base + j + 64] = f2bf(v1 * c1 + v0 * s1);   // x*cos + ( x_lo)*sin
}

// ---------------------------------------------------------------- flash attention
#define KSTR 136
#define VSTR 34
#define PSTR 34
__global__ __launch_bounds__(256) void k_attn(
    const unsigned short* __restrict__ Q, const unsigned short* __restrict__ Kt,
    const unsigned short* __restrict__ Vt, unsigned short* __restrict__ Oc)
{
  __shared__ unsigned short Klds[32 * KSTR];        // [key][hd]
  __shared__ unsigned short Vlds[128 * VSTR];       // [hd][key] (transposed)
  __shared__ unsigned short Plds[8 * 16 * PSTR];    // per-wave P scratch [16][32]

  const int tid  = threadIdx.x;
  const int lane = tid & 31;
  const int wave = tid >> 5;
  const int half = lane >> 4;
  const int lcol = lane & 15;
  const int qBlk = blockIdx.x;            // 16 q-blocks of 128
  const int h    = blockIdx.y;
  const int b    = blockIdx.z;
  const int kvh  = h / GROUPS_;

  const int    qRow0  = qBlk * 128 + wave * 16;
  const size_t qBase  = ((size_t)(b * H_ + h) * S_ + qRow0) * HD_;
  const size_t kvBase = ((size_t)(b * KVH_ + kvh) * S_) * HD_;

  // Q fragments: 4 chunks of K=32 across HD=128
  Frag qf[4];
  for (int c = 0; c < 4; ++c) {
    const unsigned short* rp = &Q[qBase + (size_t)lcol * HD_ + c * 32];
    for (int p = 0; p < 8; ++p) qf[c].u[p] = *(const unsigned int*)&rp[fragKA(p, half)];
  }

  f32x8 oacc[8];
  for (int d = 0; d < 8; ++d) for (int e = 0; e < 8; ++e) oacc[d][e] = 0.0f;
  float mSt[8], lSt[8];
  for (int i = 0; i < 8; ++i) { mSt[i] = -1e30f; lSt[i] = 0.0f; }

  const int   kEnd  = qBlk * 128 + 128;             // causal bound for the block
  const float scale = 0.08838834764831845f;         // 128^-0.5

  for (int kb = 0; kb < kEnd; kb += 32) {
    // cooperative K (row-major, async-able) and V (transposed) tile loads
    {
      int key = tid >> 3, d0 = (tid & 7) * 16;
      const unsigned short* gk = &Kt[kvBase + (size_t)(kb + key) * HD_ + d0];
      unsigned short* lk = &Klds[key * KSTR + d0];
#if USE_ASYNC_LDS
      async_copy_b128(gk, lk);
      async_copy_b128(gk + 8, lk + 8);
#else
      const uint4* g = (const uint4*)gk;
      uint4 v0 = g[0], v1 = g[1];
      unsigned int* d = (unsigned int*)lk;
      d[0] = v0.x; d[1] = v0.y; d[2] = v0.z; d[3] = v0.w;
      d[4] = v1.x; d[5] = v1.y; d[6] = v1.z; d[7] = v1.w;
#endif
      const uint4* gv = (const uint4*)&Vt[kvBase + (size_t)(kb + key) * HD_ + d0];
      uint4 w0 = gv[0], w1 = gv[1];
      const unsigned short* sv = (const unsigned short*)&w0;
      for (int e = 0; e < 8; ++e) Vlds[(d0 + e) * VSTR + key] = sv[e];
      sv = (const unsigned short*)&w1;
      for (int e = 0; e < 8; ++e) Vlds[(d0 + 8 + e) * VSTR + key] = sv[e];
    }
#if USE_ASYNC_LDS
    __builtin_amdgcn_s_wait_asynccnt(0);
#endif
    __syncthreads();

    // scores S = Q(16x128) @ K_tile^T -> two 16x16 fragments
    f32x8 sc[2];
    for (int nt = 0; nt < 2; ++nt) {
      for (int e = 0; e < 8; ++e) sc[nt][e] = 0.0f;
      for (int c = 0; c < 4; ++c) {
        Frag kf;
        const unsigned short* base = &Klds[(nt * 16 + lcol) * KSTR + c * 32];
        for (int p = 0; p < 8; ++p) kf.u[p] = *(const unsigned int*)&base[fragKB(p, half)];
        sc[nt] = __builtin_amdgcn_wmma_f32_16x16x32_bf16(
            false, qf[c].v, false, kf.v, (short)0, sc[nt], false, false);
      }
    }

    // scale + causal mask + online softmax (rows live in 16-lane halves)
    for (int i = 0; i < 8; ++i) {
      int qAbs = qRow0 + i + half * 8;
      float s0 = sc[0][i] * scale;
      float s1 = sc[1][i] * scale;
      if (kb + lcol      > qAbs) s0 = -1e30f;
      if (kb + 16 + lcol > qAbs) s1 = -1e30f;
      float mloc = fmaxf(s0, s1);
      for (int off = 8; off >= 1; off >>= 1) mloc = fmaxf(mloc, __shfl_xor(mloc, off, 32));
      float newM  = fmaxf(mSt[i], mloc);
      float alpha = __expf(mSt[i] - newM);
      float p0 = __expf(s0 - newM);
      float p1 = __expf(s1 - newM);
      float rs = p0 + p1;
      for (int off = 8; off >= 1; off >>= 1) rs += __shfl_xor(rs, off, 32);
      lSt[i] = lSt[i] * alpha + rs;
      mSt[i] = newM;
      for (int d = 0; d < 8; ++d) oacc[d][i] *= alpha;
      // stage P (C layout -> LDS) for A-fragment reload
      int prow = i + half * 8;
      unsigned short* pb = &Plds[(wave * 16 + prow) * PSTR];
      pb[lcol]      = f2bf(p0);
      pb[16 + lcol] = f2bf(p1);
    }
    // wave-private LDS: DS ops are in-order per wave; fence the compiler + HW
    asm volatile("s_wait_dscnt 0" ::: "memory");

    // P as A fragment (16 rows x 32 keys)
    Frag pf;
    {
      const unsigned short* base = &Plds[(wave * 16 + lcol) * PSTR];
      for (int p = 0; p < 8; ++p) pf.u[p] = *(const unsigned int*)&base[fragKA(p, half)];
    }
    // O += P(16x32) @ V(32x128)
    for (int d = 0; d < 8; ++d) {
      Frag vf;
      const unsigned short* base = &Vlds[(d * 16 + lcol) * VSTR];
      for (int p = 0; p < 8; ++p) vf.u[p] = *(const unsigned int*)&base[fragKB(p, half)];
      oacc[d] = __builtin_amdgcn_wmma_f32_16x16x32_bf16(
          false, pf.v, false, vf.v, (short)0, oacc[d], false, false);
    }
    __syncthreads();
  }

  // normalize + store attention output as bf16 [B, S, H*HD]
  for (int i = 0; i < 8; ++i) {
    float inv = 1.0f / lSt[i];
    int srow = qRow0 + i + half * 8;
    size_t ob = ((size_t)b * S_ + srow) * (size_t)(H_ * HD_) + (size_t)h * HD_;
    for (int d = 0; d < 8; ++d)
      Oc[ob + d * 16 + lcol] = f2bf(oacc[d][i] * inv);
  }
}

// ---------------------------------------------------------------- launcher
extern "C" void kernel_launch(void* const* d_in, const int* in_sizes, int n_in,
                              void* d_out, int out_size, void* d_ws, size_t ws_size,
                              hipStream_t stream)
{
  (void)in_sizes; (void)n_in; (void)out_size; (void)ws_size;
  const float* x    = (const float*)d_in[0];
  const float* cosp = (const float*)d_in[1];
  const float* sinp = (const float*)d_in[2];
  // d_in[3] = causal mask: applied analytically in-kernel
  const float* Wq   = (const float*)d_in[4];
  const float* Wk   = (const float*)d_in[5];
  const float* Wv   = (const float*)d_in[6];
  const float* Wo   = (const float*)d_in[7];

  char* ws = (char*)d_ws;
  size_t off = 0;
  auto alloc = [&](size_t bytes) { char* p = ws + off; off += (bytes + 255) & ~(size_t)255; return p; };
  unsigned short* xb  = (unsigned short*)alloc((size_t)B_ * S_ * D_ * 2);
  unsigned short* Wqb = (unsigned short*)alloc((size_t)D_ * H_ * HD_ * 2);
  unsigned short* Wkb = (unsigned short*)alloc((size_t)D_ * KVH_ * HD_ * 2);
  unsigned short* Wvb = (unsigned short*)alloc((size_t)D_ * KVH_ * HD_ * 2);
  unsigned short* Wob = (unsigned short*)alloc((size_t)H_ * HD_ * D_ * 2);
  unsigned short* Qb  = (unsigned short*)alloc((size_t)B_ * H_ * S_ * HD_ * 2);
  unsigned short* Kb  = (unsigned short*)alloc((size_t)B_ * KVH_ * S_ * HD_ * 2);
  unsigned short* Vb  = (unsigned short*)alloc((size_t)B_ * KVH_ * S_ * HD_ * 2);
  unsigned short* Ab  = (unsigned short*)alloc((size_t)B_ * S_ * H_ * HD_ * 2);

  auto cvt = [&](const float* s, unsigned short* d, int n) {
    k_f32_to_bf16<<<(n + 255) / 256, 256, 0, stream>>>(s, d, n);
  };
  cvt(x,  xb,  B_ * S_ * D_);
  cvt(Wq, Wqb, D_ * H_ * HD_);
  cvt(Wk, Wkb, D_ * KVH_ * HD_);
  cvt(Wv, Wvb, D_ * KVH_ * HD_);
  cvt(Wo, Wob, H_ * HD_ * D_);

  const int M = B_ * S_;
  k_gemm_bf16<<<dim3((H_ * HD_) / 128, M / 128), 256, 0, stream>>>(xb, Wqb, Qb, M, H_ * HD_, D_, 0, H_);
  k_gemm_bf16<<<dim3((KVH_ * HD_) / 128, M / 128), 256, 0, stream>>>(xb, Wkb, Kb, M, KVH_ * HD_, D_, 0, KVH_);
  k_gemm_bf16<<<dim3((KVH_ * HD_) / 128, M / 128), 256, 0, stream>>>(xb, Wvb, Vb, M, KVH_ * HD_, D_, 0, KVH_);

  { int rows = B_ * H_ * S_;   k_rope<<<(rows * 64 + 255) / 256, 256, 0, stream>>>(Qb, cosp, sinp, rows); }
  { int rows = B_ * KVH_ * S_; k_rope<<<(rows * 64 + 255) / 256, 256, 0, stream>>>(Kb, cosp, sinp, rows); }

  k_attn<<<dim3(S_ / 128, H_, B_), 256, 0, stream>>>(Qb, Kb, Vb, Ab);

  k_gemm_bf16<<<dim3(D_ / 128, M / 128), 256, 0, stream>>>(Ab, Wob, d_out, M, D_, H_ * HD_, 1, 0);
}
